// DecoderRNN_75668733820951
// MI455X (gfx1250) — matrix operand order repaired and checked by reading, not compiled, not run on hardware
//
#include <hip/hip_runtime.h>
#include <hip/hip_bf16.h>

#define EMBED  1024
#define HIDDEN 1024
#define VOCAB  32000
#define BSZ    32
#define LENC   49
#define TSTEPS 24

typedef __attribute__((ext_vector_type(16))) __bf16 v16bf;
typedef __attribute__((ext_vector_type(8)))  float  v8f;

union Frag16 {
    v16bf bf;
    struct { uint4 lo; uint4 hi; } q;
};

__device__ __forceinline__ unsigned short f2bf(float f) {
    unsigned int x = __float_as_uint(f);
    x += 0x7FFFu + ((x >> 16) & 1u);           // round-to-nearest-even
    return (unsigned short)(x >> 16);
}

__device__ __forceinline__ float sigmoidf_(float x) {
    return 1.0f / (1.0f + __expf(-x));
}

// ---- WMMA fragment loaders (bf16, wave32) -------------------------------
// A fragment: 16x32 tile, row-major source [m][k].
// lane l (m = l&15, half = l>>4): elem e -> K = half*16 + e
// => 16 contiguous elements at k = half*16.
__device__ __forceinline__ v16bf load_a_frag(const unsigned short* __restrict__ base,
                                             int ld, int kbase) {
    int lane = threadIdx.x & 31;
    int m = lane & 15, half = lane >> 4;
    const unsigned short* p = base + (size_t)m * ld + kbase + half * 16;
    Frag16 f;
    f.q.lo = *reinterpret_cast<const uint4*>(p);
    f.q.hi = *reinterpret_cast<const uint4*>(p + 8);
    return f.bf;
}

// B fragment: 32x16 tile (KxN) from W stored row-major [N][K] (W^T mathematically).
// lane l (n = l&15, half = l>>4): elem e -> K = half*16 + e  => 16 contiguous.
__device__ __forceinline__ v16bf load_b_frag(const unsigned short* __restrict__ W,
                                             int ldw, int col0, int kbase) {
    int lane = threadIdx.x & 31;
    int n = lane & 15, half = lane >> 4;
    const unsigned short* p = W + (size_t)(col0 + n) * ldw + kbase + half * 16;
    Frag16 f;
    f.q.lo = *reinterpret_cast<const uint4*>(p);
    f.q.hi = *reinterpret_cast<const uint4*>(p + 8);
    return f.bf;
}

__device__ __forceinline__ v8f wmma_bf16(v16bf a, v16bf b, v8f c) {
    return __builtin_amdgcn_wmma_f32_16x16x32_bf16(false, a, false, b,
                                                   (short)0, c, false, false);
}

// ---- preprocessing ------------------------------------------------------
__global__ __launch_bounds__(256) void f32_to_bf16_kernel(
        const float* __restrict__ src, unsigned short* __restrict__ dst, int n4) {
    int i = blockIdx.x * blockDim.x + threadIdx.x;
    int stride = gridDim.x * blockDim.x;
    for (; i < n4; i += stride) {
        float4 v = reinterpret_cast<const float4*>(src)[i];
        ushort4 o;
        o.x = f2bf(v.x); o.y = f2bf(v.y); o.z = f2bf(v.z); o.w = f2bf(v.w);
        reinterpret_cast<ushort4*>(dst)[i] = o;
    }
}

// emb_bf layout: [T][B][EMBED]
__global__ __launch_bounds__(256) void gather_embed_kernel(
        const float* __restrict__ embed_table, const int* __restrict__ captions,
        unsigned short* __restrict__ emb_bf) {
    int t = blockIdx.x / BSZ;
    int b = blockIdx.x % BSZ;
    int tok = captions[b * TSTEPS + t];
    const float* src = embed_table + (size_t)tok * EMBED;
    unsigned short* dst = emb_bf + ((size_t)t * BSZ + b) * EMBED;
    for (int e = threadIdx.x; e < EMBED; e += 256)
        dst[e] = f2bf(src[e]);
}

__global__ __launch_bounds__(256) void init_state_kernel(
        float* hf, float* cf, unsigned short* hbf0, unsigned short* hbf1) {
    int i = blockIdx.x * blockDim.x + threadIdx.x;
    if (i < BSZ * HIDDEN) {
        hf[i] = 0.0f; cf[i] = 0.0f; hbf0[i] = 0; hbf1[i] = 0;
    }
}

// ---- per-step: attention + softmax + context, builds x = [context|emb] --
__global__ __launch_bounds__(256) void attention_kernel(
        const float* __restrict__ features,      // [B][L][E] fp32
        const float* __restrict__ hf,            // [B][H] fp32
        const unsigned short* __restrict__ embt, // [B][E] bf16 (this t)
        unsigned short* __restrict__ xbf) {      // [B][2E] bf16
    __shared__ float sh[HIDDEN];
    __shared__ float sw[64];
    int b = blockIdx.x;
    const float* fb = features + (size_t)b * LENC * EMBED;

    for (int e = threadIdx.x; e < HIDDEN; e += 256) sh[e] = hf[b * HIDDEN + e];
    __syncthreads();

    int wave = threadIdx.x >> 5, lane = threadIdx.x & 31;
    for (int l = wave; l < LENC; l += 8) {
        const float* fl = fb + (size_t)l * EMBED;
        float s = 0.0f;
        for (int k = lane; k < EMBED; k += 32) s += fl[k] * sh[k];
        for (int off = 16; off > 0; off >>= 1) s += __shfl_xor(s, off, 32);
        if (lane == 0) sw[l] = s;
    }
    __syncthreads();
    if (threadIdx.x == 0) {
        float mx = -3.0e38f;
        for (int l = 0; l < LENC; ++l) mx = fmaxf(mx, sw[l]);
        float sum = 0.0f;
        for (int l = 0; l < LENC; ++l) { float e = __expf(sw[l] - mx); sw[l] = e; sum += e; }
        float inv = 1.0f / sum;
        for (int l = 0; l < LENC; ++l) sw[l] *= inv;
    }
    __syncthreads();
    for (int e = threadIdx.x; e < EMBED; e += 256) {
        float acc = 0.0f;
        for (int l = 0; l < LENC; ++l) acc += sw[l] * fb[(size_t)l * EMBED + e];
        xbf[b * 2 * EMBED + e] = f2bf(acc);
        xbf[b * 2 * EMBED + EMBED + e] = embt[b * EMBED + e];
    }
}

// ---- per-step: fused gate GEMM (WMMA) + LSTM cell update ----------------
// grid = HIDDEN/16 WGs; 4 waves, wave g computes gate g's [32 x 16] tile.
__global__ __launch_bounds__(128) void gates_lstm_kernel(
        const unsigned short* __restrict__ Wih,   // [4H][2E] bf16
        const unsigned short* __restrict__ Whh,   // [4H][H] bf16
        const unsigned short* __restrict__ xbf,   // [B][2E] bf16
        const unsigned short* __restrict__ hbf_in,// [B][H] bf16 (old h)
        const float* __restrict__ b_ih, const float* __restrict__ b_hh,
        float* __restrict__ hf, float* __restrict__ cf,
        unsigned short* __restrict__ hbf_out) {
    __shared__ float sg[4][32][16];
    int wave = threadIdx.x >> 5;                 // gate index: i,f,g,o
    int j0 = blockIdx.x * 16;                    // hidden-unit tile base
    int r0 = wave * HIDDEN + j0;                 // row base in W_ih/W_hh

    v8f acc0 = {}; v8f acc1 = {};
    // K = 0..2047 : x @ W_ih^T
    for (int ck = 0; ck < (2 * EMBED) / 32; ++ck) {
        int kb = ck * 32;
        v16bf a0 = load_a_frag(xbf, 2 * EMBED, kb);
        v16bf a1 = load_a_frag(xbf + (size_t)16 * 2 * EMBED, 2 * EMBED, kb);
        v16bf bb = load_b_frag(Wih, 2 * EMBED, r0, kb);
        acc0 = wmma_bf16(a0, bb, acc0);
        acc1 = wmma_bf16(a1, bb, acc1);
    }
    // K = 0..1023 : h @ W_hh^T
    for (int ck = 0; ck < HIDDEN / 32; ++ck) {
        int kb = ck * 32;
        v16bf a0 = load_a_frag(hbf_in, HIDDEN, kb);
        v16bf a1 = load_a_frag(hbf_in + (size_t)16 * HIDDEN, HIDDEN, kb);
        v16bf bb = load_b_frag(Whh, HIDDEN, r0, kb);
        acc0 = wmma_bf16(a0, bb, acc0);
        acc1 = wmma_bf16(a1, bb, acc1);
    }

    int lane = threadIdx.x & 31;
    int n = lane & 15, half = lane >> 4;
#pragma unroll
    for (int v = 0; v < 8; ++v) {
        int m = v + half * 8;
        sg[wave][m][n]      = acc0[v];
        sg[wave][m + 16][n] = acc1[v];
    }
    __syncthreads();

    for (int o = threadIdx.x; o < BSZ * 16; o += 128) {
        int b = o >> 4, nn = o & 15;
        int col = j0 + nn;
        float gi = sg[0][b][nn] + b_ih[col]              + b_hh[col];
        float gf = sg[1][b][nn] + b_ih[HIDDEN + col]     + b_hh[HIDDEN + col];
        float gg = sg[2][b][nn] + b_ih[2 * HIDDEN + col] + b_hh[2 * HIDDEN + col];
        float go = sg[3][b][nn] + b_ih[3 * HIDDEN + col] + b_hh[3 * HIDDEN + col];
        float cold = cf[b * HIDDEN + col];
        float cn = sigmoidf_(gf) * cold + sigmoidf_(gi) * tanhf(gg);
        float hn = sigmoidf_(go) * tanhf(cn);
        cf[b * HIDDEN + col] = cn;
        hf[b * HIDDEN + col] = hn;
        hbf_out[b * HIDDEN + col] = f2bf(hn);
    }
}

// ---- per-step: vocab projection (WMMA) ----------------------------------
// grid = VOCAB/64 WGs; 4 waves, each wave one [32 x 16] logits tile.
__global__ __launch_bounds__(128) void logits_kernel(
        const unsigned short* __restrict__ Wout,  // [V][H] bf16
        const unsigned short* __restrict__ hbf,   // [B][H] bf16 (new h)
        const float* __restrict__ b_out,
        float* __restrict__ out, int t) {
    int wave = threadIdx.x >> 5;
    int col0 = blockIdx.x * 64 + wave * 16;

    v8f acc0 = {}; v8f acc1 = {};
    for (int ck = 0; ck < HIDDEN / 32; ++ck) {
        int kb = ck * 32;
        v16bf a0 = load_a_frag(hbf, HIDDEN, kb);
        v16bf a1 = load_a_frag(hbf + (size_t)16 * HIDDEN, HIDDEN, kb);
        v16bf bb = load_b_frag(Wout, HIDDEN, col0, kb);
        acc0 = wmma_bf16(a0, bb, acc0);
        acc1 = wmma_bf16(a1, bb, acc1);
    }

    int lane = threadIdx.x & 31;
    int n = lane & 15, half = lane >> 4;
    float bias = b_out[col0 + n];
#pragma unroll
    for (int v = 0; v < 8; ++v) {
        int m = v + half * 8;
        out[((size_t)m * TSTEPS + t) * VOCAB + col0 + n]        = acc0[v] + bias;
        out[((size_t)(m + 16) * TSTEPS + t) * VOCAB + col0 + n] = acc1[v] + bias;
    }
}

// ---- host side -----------------------------------------------------------
extern "C" void kernel_launch(void* const* d_in, const int* in_sizes, int n_in,
                              void* d_out, int out_size, void* d_ws, size_t ws_size,
                              hipStream_t stream) {
    const float* features    = (const float*)d_in[0];  // [B][L][E]
    const int*   captions    = (const int*)  d_in[1];  // [B][T]
    const float* embed_table = (const float*)d_in[2];  // [V][E]
    const float* W_ih        = (const float*)d_in[3];  // [4H][2E]
    const float* W_hh        = (const float*)d_in[4];  // [4H][H]
    const float* b_ih        = (const float*)d_in[5];
    const float* b_hh        = (const float*)d_in[6];
    const float* W_out       = (const float*)d_in[7];  // [V][H]
    const float* b_out       = (const float*)d_in[8];
    float* out = (float*)d_out;

    // workspace carve (256B aligned)
    char* ws = (char*)d_ws;
    size_t off = 0;
    auto carve = [&](size_t bytes) -> char* {
        char* p = ws + off;
        off = (off + bytes + 255) & ~(size_t)255;
        return p;
    };
    unsigned short* Wih_bf  = (unsigned short*)carve((size_t)4 * HIDDEN * 2 * EMBED * 2);
    unsigned short* Whh_bf  = (unsigned short*)carve((size_t)4 * HIDDEN * HIDDEN * 2);
    unsigned short* Wout_bf = (unsigned short*)carve((size_t)VOCAB * HIDDEN * 2);
    unsigned short* emb_bf  = (unsigned short*)carve((size_t)TSTEPS * BSZ * EMBED * 2);
    unsigned short* xbf     = (unsigned short*)carve((size_t)BSZ * 2 * EMBED * 2);
    unsigned short* hbf0    = (unsigned short*)carve((size_t)BSZ * HIDDEN * 2);
    unsigned short* hbf1    = (unsigned short*)carve((size_t)BSZ * HIDDEN * 2);
    float* hf = (float*)carve((size_t)BSZ * HIDDEN * 4);
    float* cf = (float*)carve((size_t)BSZ * HIDDEN * 4);
    (void)ws_size; (void)in_sizes; (void)n_in; (void)out_size;

    // one-time (per launch) preprocessing: bf16 weight conversion + embed gather
    f32_to_bf16_kernel<<<4096, 256, 0, stream>>>(W_ih,  Wih_bf,  (4 * HIDDEN * 2 * EMBED) / 4);
    f32_to_bf16_kernel<<<2048, 256, 0, stream>>>(W_hh,  Whh_bf,  (4 * HIDDEN * HIDDEN) / 4);
    f32_to_bf16_kernel<<<8192, 256, 0, stream>>>(W_out, Wout_bf, (VOCAB * HIDDEN) / 4);
    gather_embed_kernel<<<TSTEPS * BSZ, 256, 0, stream>>>(embed_table, captions, emb_bf);
    init_state_kernel<<<(BSZ * HIDDEN + 255) / 256, 256, 0, stream>>>(hf, cf, hbf0, hbf1);

    for (int t = 0; t < TSTEPS; ++t) {
        const unsigned short* hin  = (t & 1) ? hbf1 : hbf0;
        unsigned short*       hout = (t & 1) ? hbf0 : hbf1;
        attention_kernel<<<BSZ, 256, 0, stream>>>(
            features, hf, emb_bf + (size_t)t * BSZ * EMBED, xbf);
        gates_lstm_kernel<<<HIDDEN / 16, 128, 0, stream>>>(
            Wih_bf, Whh_bf, xbf, hin, b_ih, b_hh, hf, cf, hout);
        logits_kernel<<<VOCAB / 64, 128, 0, stream>>>(
            Wout_bf, hout, b_out, out, t);
    }
}
